// DegreeEmbeddingNetwork_29927332118768
// MI455X (gfx1250) — matrix operand order repaired and checked by reading, not compiled, not run on hardware
//
#include <hip/hip_runtime.h>
#include <hip/hip_bf16.h>

typedef _Float16 v16h __attribute__((ext_vector_type(16)));
typedef _Float16 v8h  __attribute__((ext_vector_type(8)));
typedef _Float16 v4h  __attribute__((ext_vector_type(4)));
typedef float    v8f  __attribute__((ext_vector_type(8)));
typedef float    v4f  __attribute__((ext_vector_type(4)));
typedef int      i32x4 __attribute__((vector_size(16)));   // matches builtin param

namespace {
constexpr int   BE       = 64;                    // edges per tile
constexpr int   DE       = 160;                   // output embedding dim
constexpr float INV_S96  = 0.102062072615966f;    // 1/sqrt(96)
constexpr float INV_S128 = 0.0883883476483184f;   // 1/sqrt(128)
constexpr float INV_S32  = 0.176776695296637f;    // 1/sqrt(32)
}

// ---------------- gfx1250 async global->LDS helpers ----------------
#ifdef __has_builtin
#if __has_builtin(__builtin_amdgcn_global_load_async_to_lds_b128) && \
    __has_builtin(__builtin_amdgcn_global_load_async_to_lds_b32)
#define HAVE_ASYNC_LDS 1
#endif
#endif
#ifndef HAVE_ASYNC_LDS
#define HAVE_ASYNC_LDS 0
#endif

__device__ __forceinline__ void gld_async_b128(void* g, void* l) {
#if HAVE_ASYNC_LDS
  __builtin_amdgcn_global_load_async_to_lds_b128(
      (__attribute__((address_space(1))) i32x4*)g,
      (__attribute__((address_space(3))) i32x4*)l, 0, 0);
#else
  *(v4f*)l = *(const v4f*)g;
#endif
}

__device__ __forceinline__ void gld_async_b32(void* g, void* l) {
#if HAVE_ASYNC_LDS
  __builtin_amdgcn_global_load_async_to_lds_b32(
      (__attribute__((address_space(1))) int*)g,
      (__attribute__((address_space(3))) int*)l, 0, 0);
#else
  *(int*)l = *(const int*)g;
#endif
}

__device__ __forceinline__ void async_wait0() {
#if HAVE_ASYNC_LDS
#ifdef __has_builtin
#if __has_builtin(__builtin_amdgcn_s_wait_asynccnt)
  __builtin_amdgcn_s_wait_asynccnt(0);
#else
  asm volatile("s_wait_asynccnt 0x0" ::: "memory");
#endif
#else
  asm volatile("s_wait_asynccnt 0x0" ::: "memory");
#endif
#endif
}

// A fragment: 16x32 f16 (MxK) from row-major LDS (lda halves).
// lane L: M = L&15 ; elem j: K = (j>>3)*16 + (L>>4)*8 + (j&7)
__device__ __forceinline__ v16h lda_frag(const _Float16* base, int lda,
                                         int mb, int kb, int lane) {
  int row = mb + (lane & 15);
  const _Float16* p = base + row * lda + kb + (lane >> 4) * 8;
  v8h lo = *(const v8h*)p;
  v8h hi = *(const v8h*)(p + 16);
  v16h r;
#pragma unroll
  for (int i = 0; i < 8; ++i) { r[i] = lo[i]; r[i + 8] = hi[i]; }
  return r;
}

// B fragment: 32x16 f16 (KxN) from column-major LDS (ldb halves per column).
// lane L: N = L&15 ; elem j: K = (j>>3)*16 + (L>>4)*8 + (j&7)
__device__ __forceinline__ v16h ldb_frag(const _Float16* base, int ldb,
                                         int nb, int kb, int lane) {
  int col = nb + (lane & 15);
  const _Float16* p = base + col * ldb + kb + (lane >> 4) * 8;
  v8h lo = *(const v8h*)p;
  v8h hi = *(const v8h*)(p + 16);
  v16h r;
#pragma unroll
  for (int i = 0; i < 8; ++i) { r[i] = lo[i]; r[i + 8] = hi[i]; }
  return r;
}

__device__ __forceinline__ v8f wmma_k64(v16h a0, v16h a1, v16h b0, v16h b1) {
  v8f acc = {};
  acc = __builtin_amdgcn_wmma_f32_16x16x32_f16(false, a0, false, b0,
                                               (short)0, acc, false, false);
  acc = __builtin_amdgcn_wmma_f32_16x16x32_f16(false, a1, false, b1,
                                               (short)0, acc, false, false);
  return acc;
}

__global__ void zero_out_kernel(float* __restrict__ out, int n) {
  int i = blockIdx.x * blockDim.x + threadIdx.x;
  if (i < n) out[i] = 0.f;
}

__global__ __launch_bounds__(256) void deg_embed_kernel(
    const float* __restrict__ g_scalars,   // (E,64)
    const float* __restrict__ g_attr,      // (E,4)
    const int*   __restrict__ g_dst,       // (E)
    const float* __restrict__ g_lin_w,     // (64)
    const float* __restrict__ g_lin_b,     // (64)
    const float* __restrict__ g_rad_w1,    // (64,64)
    const float* __restrict__ g_rad_g,     // (64)
    const float* __restrict__ g_rad_beta,  // (64)
    const float* __restrict__ g_rad_w2,    // (64,224) -> cols 0..127 live
    const float* __restrict__ g_rad_off,   // (224)    -> 0..127 live
    const float* __restrict__ g_proj_w0,   // (96,64)  -> rows 0..63 live
    const float* __restrict__ g_proj_b0,   // (64)
    const float* __restrict__ g_proj_w1,   // (128,32) -> rows 0..63 live
    float* __restrict__ out,               // (N,160)
    int nE, int tiles) {
  // ------------- static LDS: 64,256 bytes, weights persistent -------------
  __shared__ __align__(16) _Float16 scH[BE * 64];   // scalars f16 ; later Hh
  __shared__ __align__(16) _Float16 w1c[64 * 64];   // rad_w1 col-major
  __shared__ __align__(16) _Float16 w2c[128 * 64];  // rad_w2[:, :128] col-major
  __shared__ __align__(16) _Float16 pw0[64 * 64];   // proj_w0 rows 0..63 col-major
  __shared__ __align__(16) _Float16 pw1[32 * 64];   // proj_w1 rows 0..63 col-major
  __shared__ __align__(16) float    Hf[BE * 64];    // H f32 ; later A0+B1 (f16)
  __shared__ __align__(16) float    attrT[BE * 4];  // raw src_attr tile
  __shared__ __align__(16) int      dsts[BE];
  __shared__ float s0s[64];
  __shared__ float roff[128];
  __shared__ float gln[64], bln[64], b0s[64];

  _Float16* Hh = scH;                 // reuse after GEMM1 consumed scalars
  _Float16* A0 = (_Float16*)Hf;       // reuse after LN consumed H f32
  _Float16* B1 = A0 + BE * 64;

  const int tid  = threadIdx.x;
  const int lane = tid & 31;
  const int wv   = tid >> 5;

  // ---------------- one-time weight staging (per block) -------------------
  for (int idx = tid; idx < 64 * 64; idx += 256) {       // rad_w1 -> col-major
    int k = idx >> 6, n = idx & 63;
    w1c[n * 64 + k] = (_Float16)g_rad_w1[idx];
  }
  for (int idx = tid; idx < 64 * 128; idx += 256) {      // rad_w2 cols 0..127
    int k = idx >> 7, n = idx & 127;
    w2c[n * 64 + k] = (_Float16)g_rad_w2[k * 224 + n];
  }
  for (int idx = tid; idx < 64 * 64; idx += 256) {       // proj_w0 rows 0..63
    int k = idx >> 6, n = idx & 63;
    pw0[n * 64 + k] = (_Float16)g_proj_w0[idx];
  }
  for (int idx = tid; idx < 64 * 32; idx += 256) {       // proj_w1 rows 0..63
    int k = idx >> 5, n = idx & 31;
    pw1[n * 64 + k] = (_Float16)g_proj_w1[idx];
  }
  if (tid < 64) {
    s0s[tid] = g_lin_w[tid] + g_lin_b[tid];
    gln[tid] = g_rad_g[tid];
    bln[tid] = g_rad_beta[tid];
    b0s[tid] = g_proj_b0[tid];
  } else if (tid < 192) {
    roff[tid - 64] = g_rad_off[tid - 64];
  }
  __syncthreads();

  // ---------------- grid-stride loop over edge tiles ----------------------
  for (int tile = blockIdx.x; tile < tiles; tile += gridDim.x) {
    const int eT = tile * BE;

    // ---- stage tile: scalars (f32->f16, float4 loads), attr/dst (async) ----
    for (int c = tid; c < BE * 16; c += 256) {   // 16 float4-chunks per row
      int row = c >> 4, c4 = (c & 15) * 4;
      int e = eT + row;
      v4h h4 = {};
      if (e < nE) {
        v4f f4 = *(const v4f*)(g_scalars + (size_t)e * 64 + c4);
        h4[0] = (_Float16)f4[0]; h4[1] = (_Float16)f4[1];
        h4[2] = (_Float16)f4[2]; h4[3] = (_Float16)f4[3];
      }
      *(v4h*)(scH + row * 64 + c4) = h4;
    }
    if (tid < BE) {
      int e = eT + tid;
      if (e < nE) {
        gld_async_b128((void*)(g_attr + (size_t)e * 4), attrT + tid * 4);
        gld_async_b32((void*)(g_dst + e), dsts + tid);
      } else {
        attrT[tid * 4 + 0] = 0.f; attrT[tid * 4 + 1] = 0.f;
        attrT[tid * 4 + 2] = 0.f; attrT[tid * 4 + 3] = 0.f;
        dsts[tid] = 0;
      }
    }
    async_wait0();
    __syncthreads();

    // prefetch next tile's scalars while GEMM1 runs (global_prefetch_b8)
    {
      int nextT = tile + gridDim.x;
      if (nextT < tiles) {
        size_t off = (size_t)nextT * BE * 64 + (size_t)tid * 16;
        if (off < (size_t)nE * 64) __builtin_prefetch(g_scalars + off, 0, 0);
      }
    }

    // ---- GEMM1: H = scalars @ rad_w1 (64x64 @ 64x64) ----
    {
      int mt = wv >> 1, nb = (wv & 1) * 2;
      v16h aK0 = lda_frag(scH, 64, mt * 16, 0, lane);
      v16h aK1 = lda_frag(scH, 64, mt * 16, 32, lane);
      for (int t = 0; t < 2; ++t) {
        int nt = nb + t;
        v16h bK0 = ldb_frag(w1c, 64, nt * 16, 0, lane);
        v16h bK1 = ldb_frag(w1c, 64, nt * 16, 32, lane);
        v8f acc = wmma_k64(aK0, aK1, bK0, bK1);
        int n  = nt * 16 + (lane & 15);
        int rb = mt * 16 + (lane >> 4) * 8;
#pragma unroll
        for (int r = 0; r < 8; ++r) Hf[(rb + r) * 64 + n] = acc[r];
      }
    }
    __syncthreads();

    // ---- LayerNorm + SiLU (per edge-row, f32), write Hh (over scH) ----
    if (tid < BE) {
      float sum = 0.f, sq = 0.f;
#pragma unroll 8
      for (int j = 0; j < 64; ++j) {
        float x = Hf[tid * 64 + j];
        sum += x; sq += x * x;
      }
      float mu  = sum * (1.f / 64.f);
      float var = sq * (1.f / 64.f) - mu * mu;
      float rs  = rsqrtf(var + 1e-5f);
#pragma unroll 8
      for (int j = 0; j < 64; ++j) {
        float x = (Hf[tid * 64 + j] - mu) * rs * gln[j] + bln[j];
        float y = x / (1.f + __expf(-x));          // silu
        Hh[tid * 64 + j] = (_Float16)y;
      }
    }
    __syncthreads();

    // ---- GEMM2: W = H @ rad_w2[:, :128] + fused edge transform ----
    {
      int mt = wv >> 1, nb = (wv & 1) * 4;
      v16h aK0 = lda_frag(Hh, 64, mt * 16, 0, lane);
      v16h aK1 = lda_frag(Hh, 64, mt * 16, 32, lane);
      for (int t = 0; t < 4; ++t) {
        int nt = nb + t;
        v16h bK0 = ldb_frag(w2c, 64, nt * 16, 0, lane);
        v16h bK1 = ldb_frag(w2c, 64, nt * 16, 32, lane);
        v8f acc = wmma_k64(aK0, aK1, bK0, bK1);
        int n  = nt * 16 + (lane & 15);
        int rb = mt * 16 + (lane >> 4) * 8;
#pragma unroll
        for (int r = 0; r < 8; ++r) {
          int e = rb + r;
          float v = acc[r] + roff[n];
          if (n < 64)   // w00 path: A0 = a0 * s0 * w00
            A0[e * 64 + n] = (_Float16)(attrT[e * 4] * s0s[n] * v);
          else          // w01 path: B1 = s0 * w01
            B1[e * 64 + (n - 64)] = (_Float16)(s0s[n - 64] * v);
        }
      }
    }
    __syncthreads();

    // ---- GEMM3: O0 = A0 @ proj_w0[:64] -> scatter atomics ----
    {
      int mt = wv >> 1, nb = (wv & 1) * 2;
      v16h aK0 = lda_frag(A0, 64, mt * 16, 0, lane);
      v16h aK1 = lda_frag(A0, 64, mt * 16, 32, lane);
      for (int t = 0; t < 2; ++t) {
        int nt = nb + t;
        v16h bK0 = ldb_frag(pw0, 64, nt * 16, 0, lane);
        v16h bK1 = ldb_frag(pw0, 64, nt * 16, 32, lane);
        v8f acc = wmma_k64(aK0, aK1, bK0, bK1);
        int j  = nt * 16 + (lane & 15);
        int rb = mt * 16 + (lane >> 4) * 8;
#pragma unroll
        for (int r = 0; r < 8; ++r) {
          int el = rb + r;
          if (eT + el < nE) {
            float val = (acc[r] * INV_S96 + b0s[j]) * INV_S32;
            atomicAdd(&out[(size_t)dsts[el] * DE + j], val);
          }
        }
      }
    }

    // ---- GEMM4: Q = B1 @ proj_w1[:64] ; o1 = Q x a1 -> scatter ----
    {
      int mt = wv >> 1, nt = (wv & 1);
      v16h aK0 = lda_frag(B1, 64, mt * 16, 0, lane);
      v16h aK1 = lda_frag(B1, 64, mt * 16, 32, lane);
      v16h bK0 = ldb_frag(pw1, 64, nt * 16, 0, lane);
      v16h bK1 = ldb_frag(pw1, 64, nt * 16, 32, lane);
      v8f acc = wmma_k64(aK0, aK1, bK0, bK1);
      int vc = nt * 16 + (lane & 15);
      int rb = mt * 16 + (lane >> 4) * 8;
#pragma unroll
      for (int r = 0; r < 8; ++r) {
        int el = rb + r;
        if (eT + el < nE) {
          float q = acc[r] * INV_S128 * INV_S32;
          size_t base = (size_t)dsts[el] * DE + 64 + (size_t)vc * 3;
          atomicAdd(&out[base + 0], q * attrT[el * 4 + 1]);
          atomicAdd(&out[base + 1], q * attrT[el * 4 + 2]);
          atomicAdd(&out[base + 2], q * attrT[el * 4 + 3]);
        }
      }
    }
    __syncthreads();   // protect scH/attrT/dsts/Hf before next tile staging
  }
}

extern "C" void kernel_launch(void* const* d_in, const int* in_sizes, int n_in,
                              void* d_out, int out_size, void* d_ws, size_t ws_size,
                              hipStream_t stream) {
  // setup_inputs() order:
  // 0 dst_input (shape only), 1 src_attr, 2 scalars, 3 lin_w, 4 lin_b,
  // 5 rad_w1, 6 rad_g, 7 rad_beta, 8 rad_w2, 9 rad_off,
  // 10 proj_w0, 11 proj_b0, 12 proj_w1, 13 dst_index
  const float* g_attr     = (const float*)d_in[1];
  const float* g_scalars  = (const float*)d_in[2];
  const float* g_lin_w    = (const float*)d_in[3];
  const float* g_lin_b    = (const float*)d_in[4];
  const float* g_rad_w1   = (const float*)d_in[5];
  const float* g_rad_g    = (const float*)d_in[6];
  const float* g_rad_beta = (const float*)d_in[7];
  const float* g_rad_w2   = (const float*)d_in[8];
  const float* g_rad_off  = (const float*)d_in[9];
  const float* g_proj_w0  = (const float*)d_in[10];
  const float* g_proj_b0  = (const float*)d_in[11];
  const float* g_proj_w1  = (const float*)d_in[12];
  const int*   g_dst      = (const int*)d_in[13];
  float* out = (float*)d_out;

  int nE = in_sizes[13];

  // out is accumulated with atomics: must start from zero every call.
  zero_out_kernel<<<(out_size + 255) / 256, 256, 0, stream>>>(out, out_size);

  int tiles   = (nE + BE - 1) / BE;            // 5000 for E=320000
  int nBlocks = tiles < 1280 ? tiles : 1280;   // persistent-ish grid
  deg_embed_kernel<<<nBlocks, 256, 0, stream>>>(
      g_scalars, g_attr, g_dst, g_lin_w, g_lin_b, g_rad_w1, g_rad_g,
      g_rad_beta, g_rad_w2, g_rad_off, g_proj_w0, g_proj_b0, g_proj_w1,
      out, nE, tiles);
}
